// Self_Attention_53609781789232
// MI455X (gfx1250) — matrix operand order repaired and checked by reading, not compiled, not run on hardware
//
#include <hip/hip_runtime.h>

// ---------------------------------------------------------------------------
// Self-attention, B=8 S=2048 D=1024 OUT=1024, scale = 1/sqrt(64) = 0.125
// Pipeline: cvt(x)->bf16, cvt(W)->bf16 transposed, QKV GEMM (WMMA bf16),
// scores GEMM (WMMA bf16, f32 out), row softmax, probs@V GEMM (WMMA bf16).
// GEMM inner loops are register double-buffered: next k-step's fragments are
// loaded before the current k-step's WMMAs execute.
// ---------------------------------------------------------------------------

#define BATCH 8
#define SEQ   2048
#define DIM   1024
#define OUTD  1024
#define SCALE 0.125f

typedef __attribute__((ext_vector_type(16))) __bf16          v16bf;
typedef __attribute__((ext_vector_type(8)))  float           v8f;
typedef __attribute__((ext_vector_type(8)))  unsigned short  ushort8;

union ABFrag { v16bf v; ushort8 h[2]; };

static __device__ __forceinline__ unsigned short f32_to_bf16_bits(float f) {
    unsigned int u = __float_as_uint(f);
    u += 0x7FFFu + ((u >> 16) & 1u);       // round-to-nearest-even
    return (unsigned short)(u >> 16);
}

// A-matrix 16x32 bf16 fragment (row-major source, ld in elements).
// Lane L: row = L&15; lanes 0-15 hold K 0-7 & 16-23, lanes 16-31 hold 8-15 & 24-31.
static __device__ __forceinline__ v16bf load_a_frag(const unsigned short* base,
                                                    int ld, int lane) {
    const int row  = lane & 15;
    const int koff = (lane >> 4) * 8;
    ABFrag u;
    u.h[0] = *(const ushort8*)(base + (size_t)row * ld + koff);
    u.h[1] = *(const ushort8*)(base + (size_t)row * ld + koff + 16);
    return u.v;
}

// B-matrix 32x16 bf16 fragment. Source stored "column-contiguous": row c of the
// source buffer holds column c's K values contiguously (ld elements per row).
// Lane L: col = L&15; lanes 0-15 hold K 0-7 & 16-23, lanes 16-31 hold 8-15 & 24-31.
static __device__ __forceinline__ v16bf load_b_frag(const unsigned short* base,
                                                    int ld, int lane) {
    const int col  = lane & 15;
    const int koff = (lane >> 4) * 8;
    ABFrag u;
    u.h[0] = *(const ushort8*)(base + (size_t)col * ld + koff);
    u.h[1] = *(const ushort8*)(base + (size_t)col * ld + koff + 16);
    return u.v;
}

static __device__ __forceinline__ v8f wmma_bf16(v16bf a, v16bf b, v8f c) {
    return __builtin_amdgcn_wmma_f32_16x16x32_bf16(
        /*neg_a=*/false, a, /*neg_b=*/false, b,
        /*c_mod=*/(short)0, c, /*reuse_a=*/false, /*reuse_b=*/false);
}

// ---------------------------------------------------------------------------
// Pipelined 16 (rows) x 64 (cols) x KELEMS wave GEMM tile.
// A: row-major [16, KELEMS] slab at Abase (ld = lda).
// B: four 16-col sub-tiles; column c of sub-tile t lives at
//    Bbase + (16*t + c)*ldb + k (column-contiguous layout).
// ---------------------------------------------------------------------------
template <int KELEMS>
static __device__ __forceinline__ void gemm_tile_16x64(
        const unsigned short* __restrict__ Abase, int lda,
        const unsigned short* __restrict__ Bbase, int ldb,
        int lane, v8f acc[4]) {
    v16bf a  = load_a_frag(Abase, lda, lane);
    v16bf b0 = load_b_frag(Bbase + (size_t)0  * ldb, ldb, lane);
    v16bf b1 = load_b_frag(Bbase + (size_t)16 * ldb, ldb, lane);
    v16bf b2 = load_b_frag(Bbase + (size_t)32 * ldb, ldb, lane);
    v16bf b3 = load_b_frag(Bbase + (size_t)48 * ldb, ldb, lane);

    for (int k = 32; k < KELEMS; k += 32) {
        // Prefetch the streaming A operand a few steps ahead (L2 hint).
        __builtin_prefetch(Abase + k + 256, 0, 1);
        // Issue next iteration's loads before this iteration's WMMAs.
        const v16bf an  = load_a_frag(Abase + k, lda, lane);
        const v16bf bn0 = load_b_frag(Bbase + (size_t)0  * ldb + k, ldb, lane);
        const v16bf bn1 = load_b_frag(Bbase + (size_t)16 * ldb + k, ldb, lane);
        const v16bf bn2 = load_b_frag(Bbase + (size_t)32 * ldb + k, ldb, lane);
        const v16bf bn3 = load_b_frag(Bbase + (size_t)48 * ldb + k, ldb, lane);

        acc[0] = wmma_bf16(a, b0, acc[0]);
        acc[1] = wmma_bf16(a, b1, acc[1]);
        acc[2] = wmma_bf16(a, b2, acc[2]);
        acc[3] = wmma_bf16(a, b3, acc[3]);

        a = an; b0 = bn0; b1 = bn1; b2 = bn2; b3 = bn3;
    }
    acc[0] = wmma_bf16(a, b0, acc[0]);
    acc[1] = wmma_bf16(a, b1, acc[1]);
    acc[2] = wmma_bf16(a, b2, acc[2]);
    acc[3] = wmma_bf16(a, b3, acc[3]);
}

// ---------------------------------------------------------------------------
// Kernel 1: x (f32) -> Xb (bf16), identical flat layout [B*S, D]
// ---------------------------------------------------------------------------
__global__ __launch_bounds__(256) void cvt_x_kernel(const float* __restrict__ x,
                                                    unsigned short* __restrict__ xb) {
    const size_t i = (size_t)blockIdx.x * 256 + threadIdx.x;
    xb[i] = f32_to_bf16_bits(x[i]);
}

// ---------------------------------------------------------------------------
// Kernel 2: kernel [3, D, OUT] f32 -> Wt [3, OUT, D] bf16 (transposed)
// ---------------------------------------------------------------------------
__global__ __launch_bounds__(256) void cvt_w_kernel(const float* __restrict__ w,
                                                    unsigned short* __restrict__ wt) {
    const size_t i = (size_t)blockIdx.x * 256 + threadIdx.x; // enumerates (m, out, d)
    const int d   = (int)(i % DIM);
    const int out = (int)((i / DIM) % OUTD);
    const int m   = (int)(i / ((size_t)DIM * OUTD));
    wt[i] = f32_to_bf16_bits(w[((size_t)m * DIM + d) * OUTD + out]);
}

// ---------------------------------------------------------------------------
// Kernel 3: QKV projection.  Xb [BS, D] @ Wt[m]^T -> Q/K row-major bf16,
// V stored transposed as Vt [B, OUT, S] bf16.
// Block: 128 threads (4 waves). Wave tile: 16 rows x 64 cols.
// Grid: (BS/64, 3*OUT/64)
// ---------------------------------------------------------------------------
__global__ __launch_bounds__(128) void qkv_gemm_kernel(
        const unsigned short* __restrict__ Xb,
        const unsigned short* __restrict__ Wt,
        unsigned short* __restrict__ Q,
        unsigned short* __restrict__ K,
        unsigned short* __restrict__ Vt) {
    const int lane = threadIdx.x & 31;
    const int wave = threadIdx.x >> 5;
    const int m0   = blockIdx.x * 64 + wave * 16;     // row tile in [0, B*S)
    const int nc   = blockIdx.y * 64;                 // col in [0, 3*OUT)
    const int mtx  = nc >> 10;                        // 0=Q 1=K 2=V
    const int o0   = nc & (OUTD - 1);

    const unsigned short* Abase = Xb + (size_t)m0 * DIM;
    const unsigned short* Bbase = Wt + ((size_t)mtx * OUTD + o0) * DIM;

    v8f acc[4] = {v8f{}, v8f{}, v8f{}, v8f{}};
    gemm_tile_16x64<DIM>(Abase, DIM, Bbase, DIM, lane, acc);

    const int col = lane & 15;
    const int hi  = lane >> 4;
    if (mtx < 2) {
        unsigned short* Out = (mtx == 0) ? Q : K;
#pragma unroll
        for (int t = 0; t < 4; ++t) {
            const int o = o0 + 16 * t + col;
#pragma unroll
            for (int r = 0; r < 8; ++r) {
                const int row = m0 + 8 * hi + r;
                Out[(size_t)row * OUTD + o] = f32_to_bf16_bits(acc[t][r]);
            }
        }
    } else {
        const int b     = m0 >> 11;                   // /SEQ
        const int sbase = (m0 & (SEQ - 1)) + 8 * hi;  // 8 consecutive rows
#pragma unroll
        for (int t = 0; t < 4; ++t) {
            const int o = o0 + 16 * t + col;
            ushort8 pk;
#pragma unroll
            for (int r = 0; r < 8; ++r) pk[r] = f32_to_bf16_bits(acc[t][r]);
            *(ushort8*)(Vt + ((size_t)(b * OUTD + o)) * SEQ + sbase) = pk;
        }
    }
}

// ---------------------------------------------------------------------------
// Kernel 4: scores = scale * Q @ K^T  -> f32 [B, S, S]
// K row-major is already column-contiguous for the B-operand (key n's features
// are consecutive). Block: 256 threads (8 waves), wave tile 16q x 64k.
// Grid: (BS/16, S/512)
// ---------------------------------------------------------------------------
__global__ __launch_bounds__(256) void scores_gemm_kernel(
        const unsigned short* __restrict__ Q,
        const unsigned short* __restrict__ K,
        float* __restrict__ Sc) {
    const int lane = threadIdx.x & 31;
    const int wave = threadIdx.x >> 5;
    const int q0   = blockIdx.x * 16;                 // global row in [0, B*S)
    const int b    = q0 >> 11;                        // /SEQ
    const int k0w  = blockIdx.y * 512 + wave * 64;

    const unsigned short* Abase = Q + (size_t)q0 * OUTD;
    const unsigned short* Bbase = K + ((size_t)b * SEQ + k0w) * OUTD;

    v8f acc[4] = {v8f{}, v8f{}, v8f{}, v8f{}};
    gemm_tile_16x64<OUTD>(Abase, OUTD, Bbase, OUTD, lane, acc);

    const int col = lane & 15;
    const int hi  = lane >> 4;
#pragma unroll
    for (int t = 0; t < 4; ++t) {
        const int key = k0w + 16 * t + col;
#pragma unroll
        for (int r = 0; r < 8; ++r) {
            const size_t row = (size_t)(q0 + 8 * hi + r);   // (b*S + q)
            Sc[row * SEQ + key] = acc[t][r] * SCALE;
        }
    }
}

// ---------------------------------------------------------------------------
// Kernel 5: row softmax over f32 scores -> bf16 probs [B, S, S]
// One 256-thread block per row (2048 elements, 8 per thread).
// ---------------------------------------------------------------------------
__global__ __launch_bounds__(256) void softmax_kernel(const float* __restrict__ Sc,
                                                      unsigned short* __restrict__ P) {
    const size_t row = blockIdx.x;
    const float* src = Sc + row * SEQ;
    unsigned short* dst = P + row * SEQ;
    const int t = threadIdx.x;

    float e[8];
    float m = -3.0e38f;
#pragma unroll
    for (int j = 0; j < 8; ++j) { e[j] = src[t + j * 256]; m = fmaxf(m, e[j]); }

    __shared__ float red[256];
    red[t] = m;
    __syncthreads();
    for (int s = 128; s > 0; s >>= 1) {
        if (t < s) red[t] = fmaxf(red[t], red[t + s]);
        __syncthreads();
    }
    const float rowmax = red[0];
    __syncthreads();

    float sum = 0.0f;
#pragma unroll
    for (int j = 0; j < 8; ++j) { e[j] = __expf(e[j] - rowmax); sum += e[j]; }
    red[t] = sum;
    __syncthreads();
    for (int s = 128; s > 0; s >>= 1) {
        if (t < s) red[t] += red[t + s];
        __syncthreads();
    }
    const float inv = 1.0f / red[0];
#pragma unroll
    for (int j = 0; j < 8; ++j) dst[t + j * 256] = f32_to_bf16_bits(e[j] * inv);
}

// ---------------------------------------------------------------------------
// Kernel 6: out = P @ V using Vt [B, OUT, S] (column-contiguous B operand).
// Block: 256 threads (8 waves), wave tile 16q x 64o. Grid: (BS/16, OUT/512)
// ---------------------------------------------------------------------------
__global__ __launch_bounds__(256) void out_gemm_kernel(
        const unsigned short* __restrict__ P,
        const unsigned short* __restrict__ Vt,
        float* __restrict__ Out) {
    const int lane = threadIdx.x & 31;
    const int wave = threadIdx.x >> 5;
    const int q0   = blockIdx.x * 16;                 // global row in [0, B*S)
    const int b    = q0 >> 11;
    const int o0w  = blockIdx.y * 512 + wave * 64;

    const unsigned short* Abase = P + (size_t)q0 * SEQ;
    const unsigned short* Bbase = Vt + ((size_t)b * OUTD + o0w) * SEQ;

    v8f acc[4] = {v8f{}, v8f{}, v8f{}, v8f{}};
    gemm_tile_16x64<SEQ>(Abase, SEQ, Bbase, SEQ, lane, acc);

    const int col = lane & 15;
    const int hi  = lane >> 4;
#pragma unroll
    for (int t = 0; t < 4; ++t) {
        const int o = o0w + 16 * t + col;
#pragma unroll
        for (int r = 0; r < 8; ++r) {
            const size_t row = (size_t)(q0 + 8 * hi + r);
            Out[row * OUTD + o] = acc[t][r];
        }
    }
}

// ---------------------------------------------------------------------------
// Launch
// ---------------------------------------------------------------------------
extern "C" void kernel_launch(void* const* d_in, const int* in_sizes, int n_in,
                              void* d_out, int out_size, void* d_ws, size_t ws_size,
                              hipStream_t stream) {
    const float* x = (const float*)d_in[0];      // [B, S, D]
    const float* w = (const float*)d_in[1];      // [3, D, OUT]
    float* out = (float*)d_out;                  // [B, S, OUT]
    char* ws = (char*)d_ws;

    const size_t BS = (size_t)BATCH * SEQ;

    // Workspace layout (bytes)
    const size_t off_Xb = 0;                                   // BS*D bf16   = 32 MB
    const size_t off_Wt = off_Xb + BS * DIM * 2;               // 3*OUT*D bf16=  6 MB
    const size_t off_Q  = off_Wt + (size_t)3 * OUTD * DIM * 2; // BS*OUT bf16 = 32 MB
    const size_t off_K  = off_Q  + BS * OUTD * 2;              // 32 MB
    const size_t off_Vt = off_K  + BS * OUTD * 2;              // 32 MB
    const size_t off_Sc = off_Vt + BS * OUTD * 2;              // BS*S f32    = 128 MB
    const size_t off_P  = off_Sc + BS * SEQ * 4;               // BS*S bf16   =  64 MB

    unsigned short* Xb = (unsigned short*)(ws + off_Xb);
    unsigned short* Wt = (unsigned short*)(ws + off_Wt);
    unsigned short* Q  = (unsigned short*)(ws + off_Q);
    unsigned short* K  = (unsigned short*)(ws + off_K);
    unsigned short* Vt = (unsigned short*)(ws + off_Vt);
    float*          Sc = (float*)         (ws + off_Sc);
    unsigned short* P  = (unsigned short*)(ws + off_P);

    // 1) convert x
    cvt_x_kernel<<<(unsigned)(BS * DIM / 256), 256, 0, stream>>>(x, Xb);
    // 2) convert + transpose weights
    cvt_w_kernel<<<(unsigned)((size_t)3 * DIM * OUTD / 256), 256, 0, stream>>>(w, Wt);
    // 3) QKV projections
    qkv_gemm_kernel<<<dim3((unsigned)(BS / 64), 3 * OUTD / 64), 128, 0, stream>>>(
        Xb, Wt, Q, K, Vt);
    // 4) scores
    scores_gemm_kernel<<<dim3((unsigned)(BS / 16), SEQ / 512), 256, 0, stream>>>(Q, K, Sc);
    // 5) softmax
    softmax_kernel<<<(unsigned)BS, 256, 0, stream>>>(Sc, P);
    // 6) probs @ V
    out_gemm_kernel<<<dim3((unsigned)(BS / 16), OUTD / 512), 256, 0, stream>>>(P, Vt, out);
}
